// ClusterGCN_77833397338551
// MI455X (gfx1250) — matrix operand order repaired and checked by reading, not compiled, not run on hardware
//
#include <hip/hip_runtime.h>

// ---------------------------------------------------------------------------
// Types for CDNA5 WMMA
// ---------------------------------------------------------------------------
typedef __attribute__((ext_vector_type(16))) __bf16 vbf16;
typedef __attribute__((ext_vector_type(8)))  __bf16 v8bf;
typedef __attribute__((ext_vector_type(8)))  float  v8f;

#define BN_EPS 1e-5f
#define USE_ASYNC_LDS 1

// ---------------------------------------------------------------------------
// CDNA5 async global->LDS copy (ASYNCcnt path). One instruction moves 16 B
// per lane straight into LDS without transiting VGPRs.
//   vdst = per-lane LDS byte offset (low 32 bits of generic shared pointer,
//          per the ISA LDS-aperture truncation rule)
//   vaddr = 64-bit global address (GV mode)
// ---------------------------------------------------------------------------
__device__ __forceinline__ void async_copy16(const void* g, void* l) {
#if USE_ASYNC_LDS
    unsigned lds_off = (unsigned)(size_t)l;
    unsigned long long ga = (unsigned long long)(size_t)g;
    asm volatile("global_load_async_to_lds_b128 %0, %1, off"
                 :: "v"(lds_off), "v"(ga)
                 : "memory");
#else
    *(uint4*)l = *(const uint4*)g;
#endif
}

__device__ __forceinline__ void wait_async0() {
#if USE_ASYNC_LDS
    asm volatile("s_wait_asynccnt 0x0" ::: "memory");
#endif
}

// ---------------------------------------------------------------------------
// Utility kernels
// ---------------------------------------------------------------------------
__global__ void zero_f32_kernel(float* __restrict__ p, long long n) {
    long long i = (long long)blockIdx.x * blockDim.x + threadIdx.x;
    long long stride = (long long)gridDim.x * blockDim.x;
    for (; i < n; i += stride) p[i] = 0.0f;
}

__global__ void degree_kernel(const long long* __restrict__ ei,
                              float* __restrict__ deg, int E) {
    int e = blockIdx.x * blockDim.x + threadIdx.x;
    if (e < E) atomicAdd(&deg[(int)ei[E + e]], 1.0f);
}

__global__ void invdeg_kernel(const float* __restrict__ deg,
                              float* __restrict__ invdeg, int n) {
    int i = blockIdx.x * blockDim.x + threadIdx.x;
    if (i < n) invdeg[i] = 1.0f / fmaxf(deg[i], 1.0f);
}

__global__ void cvt_bf16_kernel(const float* __restrict__ in,
                                __bf16* __restrict__ out, long long n) {
    long long i = (long long)blockIdx.x * blockDim.x + threadIdx.x;
    long long stride = (long long)gridDim.x * blockDim.x;
    for (; i < n; i += stride) out[i] = (__bf16)in[i];
}

// Convert an fp32 weight matrix W[K][NOUT] (row-major) into bf16, pre-swizzled
// into the CDNA5 WMMA B-fragment layout so the GEMM's LDS fill is a straight
// contiguous copy. Layout per 64-column block `cb`:
//   out[cb*K*64 + (kb*4+nb)*512 + lane*16 + j2] =
//     W[kb*32 + (j2>>3)*16 + (lane>>4)*8 + (j2&7)][cb*64 + nb*16 + (lane&15)]
__global__ void swz_weight_kernel(const float* __restrict__ W,
                                  __bf16* __restrict__ out, int K, int NOUT) {
    int idx = blockIdx.x * blockDim.x + threadIdx.x;
    int total = K * NOUT;
    if (idx >= total) return;
    int per  = K * 64;
    int cb   = idx / per;
    int r    = idx - cb * per;
    int frag = r >> 9;          // 512 bf16 per 32x16 fragment
    int lane = (r >> 4) & 31;
    int j2   = r & 15;
    int kb   = frag >> 2;
    int nb   = frag & 3;
    int k = kb * 32 + ((j2 >> 3) << 4) + ((lane >> 4) << 3) + (j2 & 7);
    int n = cb * 64 + (nb << 4) + (lane & 15);
    out[idx] = (__bf16)W[(size_t)k * NOUT + n];
}

// ---------------------------------------------------------------------------
// Edge scatter: agg[dst] += feat[src] (fp32 atomics, bf16 feature gather).
// One thread handles 8 consecutive channels of one edge; consecutive threads
// cover consecutive channel-chunks of the same edge (coalesced atomics).
// ---------------------------------------------------------------------------
__global__ void scatter_kernel(const __bf16* __restrict__ feat,
                               const long long* __restrict__ ei,
                               float* __restrict__ agg,
                               int E, int C, int lgch) {
    long long tid = (long long)blockIdx.x * blockDim.x + threadIdx.x;
    long long e = tid >> lgch;
    if (e >= E) return;
    int ch = (int)(tid & ((1 << lgch) - 1)) << 3;
    long long s = ei[e];
    long long d = ei[E + e];
    const v8bf v = *(const v8bf*)(feat + s * C + ch);
    float* ap = agg + d * C + ch;
#pragma unroll
    for (int j = 0; j < 8; ++j) atomicAdd(ap + j, (float)v[j]);
}

// ---------------------------------------------------------------------------
// Fused SAGE GEMM:  out = act( BN( (agg*invdeg) @ WL + bl + Xprev @ WR ) )
// Wave tile: 16 rows x 64 cols (4 x 16x16 f32 accumulators).
// Block: 8 waves = 128 rows x 64 cols; grid.y covers NOUT/64 column blocks.
// WL/WR column slices staged in LDS (async global->LDS) in pre-swizzled
// B-fragment layout; B fragments double-buffered in registers so dscnt waits
// overlap the fp32->bf16 A-side conversion VALU work.
// ---------------------------------------------------------------------------
template<int K, int NOUT, bool FUSE_BN, bool OUT_F32>
__global__ __launch_bounds__(256) void sage_gemm_kernel(
    const float*  __restrict__ agg,     // [N, K] fp32 (neighbor sums)
    const float*  __restrict__ invdeg,  // [N]
    const __bf16* __restrict__ xprev,   // [N, K] bf16 (self features)
    const __bf16* __restrict__ wl_sw,   // [NOUT/64, K*64] swizzled bf16
    const __bf16* __restrict__ wr_sw,   // [NOUT/64, K*64] swizzled bf16
    const float*  __restrict__ blias,   // [NOUT]
    const float*  __restrict__ bn_w, const float* __restrict__ bn_b,
    const float*  __restrict__ bn_m, const float* __restrict__ bn_v,
    void* __restrict__ outp, int nwavetiles)
{
    __shared__ __align__(16) __bf16 ldsWL[K * 64];
    __shared__ __align__(16) __bf16 ldsWR[K * 64];

    const int cb = blockIdx.y;                 // 64-column block
    // Async global->LDS staging of the pre-swizzled weight slices.
    {
        const char* gwl = (const char*)(wl_sw + (size_t)cb * K * 64);
        const char* gwr = (const char*)(wr_sw + (size_t)cb * K * 64);
        char* swl = (char*)ldsWL;
        char* swr = (char*)ldsWR;
        const int n16 = K * 64 / 8;            // 16-byte chunks
        for (int i = threadIdx.x; i < n16; i += 256) {
            async_copy16(gwl + i * 16, swl + i * 16);
            async_copy16(gwr + i * 16, swr + i * 16);
        }
        wait_async0();                         // own ASYNCcnt drained
    }
    __syncthreads();                           // all waves' fills visible

    const int wave = threadIdx.x >> 5;
    const int lane = threadIdx.x & 31;
    const int rt   = blockIdx.x * 8 + wave;    // 16-row tile index
    if (rt >= nwavetiles) return;              // wave-uniform: EXEC stays full

    // A-fragment row for this lane (ISA: 16-bit A 16x32, M = lane&15,
    // K-octet selected by lane>>4).
    const int mrow = rt * 16 + (lane & 15);
    const int kh   = lane >> 4;
    const float idg = invdeg[mrow];

    const float*  ab = agg   + (size_t)mrow * K + kh * 8;
    const __bf16* xb = xprev + (size_t)mrow * K + kh * 8;

    v8f acc[4];
#pragma unroll
    for (int nb = 0; nb < 4; ++nb)
#pragma unroll
        for (int i = 0; i < 8; ++i) acc[nb][i] = 0.0f;

#pragma unroll
    for (int k0 = 0; k0 < K; k0 += 32) {
        const int fragrow = (k0 >> 5) << 2;    // kb*4

        // Issue all 8 B-fragment LDS loads first (distinct registers) so the
        // dscnt waits overlap the A-side global loads and cvt VALU below.
        vbf16 bwl[4], bwr[4];
#pragma unroll
        for (int nb = 0; nb < 4; ++nb) {
            bwl[nb] = *(const vbf16*)(ldsWL + (((fragrow + nb) << 9) + (lane << 4)));
            bwr[nb] = *(const vbf16*)(ldsWR + (((fragrow + nb) << 9) + (lane << 4)));
        }

        // --- A fragment for mean = agg * invdeg (fp32 -> bf16 on the fly) ---
        float4 f0 = *(const float4*)(ab + k0);
        float4 f1 = *(const float4*)(ab + k0 + 4);
        float4 f2 = *(const float4*)(ab + k0 + 16);
        float4 f3 = *(const float4*)(ab + k0 + 20);
        vbf16 am;
        am[0]  = (__bf16)(f0.x * idg); am[1]  = (__bf16)(f0.y * idg);
        am[2]  = (__bf16)(f0.z * idg); am[3]  = (__bf16)(f0.w * idg);
        am[4]  = (__bf16)(f1.x * idg); am[5]  = (__bf16)(f1.y * idg);
        am[6]  = (__bf16)(f1.z * idg); am[7]  = (__bf16)(f1.w * idg);
        am[8]  = (__bf16)(f2.x * idg); am[9]  = (__bf16)(f2.y * idg);
        am[10] = (__bf16)(f2.z * idg); am[11] = (__bf16)(f2.w * idg);
        am[12] = (__bf16)(f3.x * idg); am[13] = (__bf16)(f3.y * idg);
        am[14] = (__bf16)(f3.z * idg); am[15] = (__bf16)(f3.w * idg);

        // --- A fragment for self features (already bf16) ---
        v8bf x0 = *(const v8bf*)(xb + k0);
        v8bf x1 = *(const v8bf*)(xb + k0 + 16);
        vbf16 ax;
#pragma unroll
        for (int j = 0; j < 8; ++j) { ax[j] = x0[j]; ax[8 + j] = x1[j]; }

        // --- 8 WMMAs for this K-step: mean x WL and self x WR, same acc ---
#pragma unroll
        for (int nb = 0; nb < 4; ++nb) {
            acc[nb] = __builtin_amdgcn_wmma_f32_16x16x32_bf16(
                false, am, false, bwl[nb], (short)0, acc[nb], false, false);
            acc[nb] = __builtin_amdgcn_wmma_f32_16x16x32_bf16(
                false, ax, false, bwr[nb], (short)0, acc[nb], false, false);
        }
    }

    // Epilogue: bias (+ BN + ReLU), store.
    // C/D layout: lanes 0-15 -> N=lane, M=i; lanes 16-31 -> N=lane-16, M=8+i.
    const int orow0 = rt * 16 + kh * 8;
#pragma unroll
    for (int nb = 0; nb < 4; ++nb) {
        const int col = cb * 64 + (nb << 4) + (lane & 15);
        const float add = blias[col];
        float scale = 1.0f, shift = 0.0f;
        if (FUSE_BN) {
            const float s = bn_w[col] * rsqrtf(bn_v[col] + BN_EPS);
            scale = s;
            shift = bn_b[col] - bn_m[col] * s;
        }
#pragma unroll
        for (int i = 0; i < 8; ++i) {
            float y = acc[nb][i] + add;
            if (FUSE_BN) y = fmaxf(y * scale + shift, 0.0f);
            const size_t o = (size_t)(orow0 + i) * NOUT + col;
            if (OUT_F32) ((float*)outp)[o] = y;
            else         ((__bf16*)outp)[o] = (__bf16)y;
        }
    }
}

// ---------------------------------------------------------------------------
// Host-side orchestration
// ---------------------------------------------------------------------------
extern "C" void kernel_launch(void* const* d_in, const int* in_sizes, int n_in,
                              void* d_out, int out_size, void* d_ws, size_t ws_size,
                              hipStream_t stream) {
    const float*     x    = (const float*)d_in[0];
    const long long* ei   = (const long long*)d_in[1];
    const float* wl1 = (const float*)d_in[2];
    const float* bl1 = (const float*)d_in[3];
    const float* wr1 = (const float*)d_in[4];
    const float* wl2 = (const float*)d_in[5];
    const float* bl2 = (const float*)d_in[6];
    const float* wr2 = (const float*)d_in[7];
    const float* wl3 = (const float*)d_in[8];
    const float* bl3 = (const float*)d_in[9];
    const float* wr3 = (const float*)d_in[10];
    const float* bn1_w = (const float*)d_in[11];
    const float* bn1_b = (const float*)d_in[12];
    const float* bn1_m = (const float*)d_in[13];
    const float* bn1_v = (const float*)d_in[14];
    const float* bn2_w = (const float*)d_in[15];
    const float* bn2_b = (const float*)d_in[16];
    const float* bn2_m = (const float*)d_in[17];
    const float* bn2_v = (const float*)d_in[18];

    const int N = in_sizes[0] / 128;
    const int E = in_sizes[1] / 2;
    const int IN_C = 128, HID_C = 256, OUT_C = 128;

    // Workspace carve-out (256 B aligned).
    char* ws = (char*)d_ws;
    size_t off = 0;
    auto take = [&](size_t bytes) -> void* {
        void* p = ws + off;
        off = (off + bytes + 255) & ~(size_t)255;
        return p;
    };
    float*  deg    = (float*)take((size_t)N * 4);
    float*  invdeg = (float*)take((size_t)N * 4);
    float*  agg    = (float*)take((size_t)N * HID_C * 4);     // max width
    __bf16* xb     = (__bf16*)take((size_t)N * IN_C * 2);
    __bf16* h1     = (__bf16*)take((size_t)N * HID_C * 2);
    __bf16* h2     = (__bf16*)take((size_t)N * HID_C * 2);
    __bf16* wl1s   = (__bf16*)take((size_t)IN_C  * HID_C * 2);
    __bf16* wr1s   = (__bf16*)take((size_t)IN_C  * HID_C * 2);
    __bf16* wl2s   = (__bf16*)take((size_t)HID_C * HID_C * 2);
    __bf16* wr2s   = (__bf16*)take((size_t)HID_C * HID_C * 2);
    __bf16* wl3s   = (__bf16*)take((size_t)HID_C * OUT_C * 2);
    __bf16* wr3s   = (__bf16*)take((size_t)HID_C * OUT_C * 2);
    (void)ws_size; (void)n_in; (void)out_size;

    const int TPB = 256;
    const int ntiles = N / 16;                     // 50000/16 = 3125
    const int gemm_gx = (ntiles + 7) / 8;

    // ---- one-time per call: degree / invdeg / feature+weight conversion ----
    zero_f32_kernel<<<1024, TPB, 0, stream>>>(deg, (long long)N);
    degree_kernel<<<(E + TPB - 1) / TPB, TPB, 0, stream>>>(ei, deg, E);
    invdeg_kernel<<<(N + TPB - 1) / TPB, TPB, 0, stream>>>(deg, invdeg, N);
    cvt_bf16_kernel<<<2048, TPB, 0, stream>>>(x, xb, (long long)N * IN_C);
    swz_weight_kernel<<<(IN_C * HID_C + TPB - 1) / TPB, TPB, 0, stream>>>(wl1, wl1s, IN_C, HID_C);
    swz_weight_kernel<<<(IN_C * HID_C + TPB - 1) / TPB, TPB, 0, stream>>>(wr1, wr1s, IN_C, HID_C);
    swz_weight_kernel<<<(HID_C * HID_C + TPB - 1) / TPB, TPB, 0, stream>>>(wl2, wl2s, HID_C, HID_C);
    swz_weight_kernel<<<(HID_C * HID_C + TPB - 1) / TPB, TPB, 0, stream>>>(wr2, wr2s, HID_C, HID_C);
    swz_weight_kernel<<<(HID_C * OUT_C + TPB - 1) / TPB, TPB, 0, stream>>>(wl3, wl3s, HID_C, OUT_C);
    swz_weight_kernel<<<(HID_C * OUT_C + TPB - 1) / TPB, TPB, 0, stream>>>(wr3, wr3s, HID_C, OUT_C);

    // ---- layer 1: SAGE(128->256) + BN + ReLU ----
    zero_f32_kernel<<<2048, TPB, 0, stream>>>(agg, (long long)N * IN_C);
    {
        long long tot = (long long)E * (IN_C / 8);
        scatter_kernel<<<(unsigned)((tot + TPB - 1) / TPB), TPB, 0, stream>>>(
            xb, ei, agg, E, IN_C, 4);
    }
    sage_gemm_kernel<128, 256, true, false><<<dim3(gemm_gx, 4), TPB, 0, stream>>>(
        agg, invdeg, xb, wl1s, wr1s, bl1, bn1_w, bn1_b, bn1_m, bn1_v, h1, ntiles);

    // ---- layer 2: SAGE(256->256) + BN + ReLU ----
    zero_f32_kernel<<<2048, TPB, 0, stream>>>(agg, (long long)N * HID_C);
    {
        long long tot = (long long)E * (HID_C / 8);
        scatter_kernel<<<(unsigned)((tot + TPB - 1) / TPB), TPB, 0, stream>>>(
            h1, ei, agg, E, HID_C, 5);
    }
    sage_gemm_kernel<256, 256, true, false><<<dim3(gemm_gx, 4), TPB, 0, stream>>>(
        agg, invdeg, h1, wl2s, wr2s, bl2, bn2_w, bn2_b, bn2_m, bn2_v, h2, ntiles);

    // ---- layer 3: SAGE(256->128), fp32 out ----
    zero_f32_kernel<<<2048, TPB, 0, stream>>>(agg, (long long)N * HID_C);
    {
        long long tot = (long long)E * (HID_C / 8);
        scatter_kernel<<<(unsigned)((tot + TPB - 1) / TPB), TPB, 0, stream>>>(
            h2, ei, agg, E, HID_C, 5);
    }
    sage_gemm_kernel<256, 128, false, true><<<dim3(gemm_gx, 2), TPB, 0, stream>>>(
        agg, invdeg, h2, wl3s, wr3s, bl3, bn2_w, bn2_b, bn2_m, bn2_v, d_out, ntiles);
}
